// MaskedAttentionBlock_63058709840623
// MI455X (gfx1250) — matrix operand order repaired and checked by reading, not compiled, not run on hardware
//
#include <hip/hip_runtime.h>
#include <hip/hip_bf16.h>

typedef __attribute__((ext_vector_type(16))) __bf16 v16bf;
typedef __attribute__((ext_vector_type(8)))  __bf16 v8bf;
typedef __attribute__((ext_vector_type(8)))  float  v8f;

#define B_DIM  2
#define S_DIM  2048
#define D_DIM  512
#define H_DIM  8
#define HD_DIM 64
#define D4_DIM 2048

// ---------- helpers ----------
__device__ __forceinline__ __bf16 f2bf(float f) {
  union { float f; unsigned u; } x; x.f = f;
  unsigned r = x.u + 0x7fffu + ((x.u >> 16) & 1u);
  unsigned short h = (unsigned short)(r >> 16);
  return __builtin_bit_cast(__bf16, h);
}
__device__ __forceinline__ v8bf ld8(const __bf16* p) { return *(const v8bf*)p; }
__device__ __forceinline__ v16bf cat8(v8bf a, v8bf b) {
  return __builtin_shufflevector(a, b, 0,1,2,3,4,5,6,7,8,9,10,11,12,13,14,15);
}

// ---------- fp32 -> bf16 conversion (weights) ----------
__global__ void cvt_f32_bf16(const float* __restrict__ in, __bf16* __restrict__ out, int n) {
  for (int i = blockIdx.x * blockDim.x + threadIdx.x; i < n; i += gridDim.x * blockDim.x)
    out[i] = f2bf(in[i]);
}

// ---------- LayerNorm (one block of 128 threads per row), bf16 out ----------
__global__ void layernorm_to_bf16(const float* __restrict__ x, const float* __restrict__ g,
                                  const float* __restrict__ bta, __bf16* __restrict__ out) {
  const int row = blockIdx.x;
  const float* xr = x + (size_t)row * D_DIM;
  __shared__ float s1[128], s2[128];
  float a = 0.f, b = 0.f;
  for (int i = threadIdx.x; i < D_DIM; i += 128) { float v = xr[i]; a += v; b += v * v; }
  s1[threadIdx.x] = a; s2[threadIdx.x] = b;
  __syncthreads();
  for (int st = 64; st > 0; st >>= 1) {
    if ((int)threadIdx.x < st) { s1[threadIdx.x] += s1[threadIdx.x + st];
                                 s2[threadIdx.x] += s2[threadIdx.x + st]; }
    __syncthreads();
  }
  const float mu  = s1[0] * (1.0f / D_DIM);
  const float var = s2[0] * (1.0f / D_DIM) - mu * mu;
  const float rs  = rsqrtf(var + 1e-5f);
  for (int i = threadIdx.x; i < D_DIM; i += 128)
    out[(size_t)row * D_DIM + i] = f2bf((xr[i] - mu) * rs * g[i] + bta[i]);
}

// ---------- bf16 WMMA GEMM: C[M,N] = A[M,K] @ B[K,N] + bias ----------
// Block = 256 threads (8 waves). Block tile: 16(M) x 512(N); wave tile 16 x 64.
// A tile (16x32) staged once per block; B tile (32x512) staged TRANSPOSED ([n][k])
// via per-thread 8x8 register transpose -> fragment reads are contiguous ds_load_b128.
// mode 0: fp32 out (+optional fp32 residual)
// mode 1: bf16 out, exact GELU
// mode 2: bf16 out
// mode 3: bf16 out scaled, scattered to [B,H,S,HD] head layout (for Q/K/V)
__global__ void gemm_bf16_wmma(const __bf16* __restrict__ A, const __bf16* __restrict__ Bm,
                               const float* __restrict__ bias,
                               float* __restrict__ Cf, __bf16* __restrict__ Cb,
                               const float* __restrict__ res,
                               int M, int N, int K, int mode, float scale) {
  __shared__ __align__(16) __bf16 ldsA[16 * 32];      // [m][k]
  __shared__ __align__(16) __bf16 ldsB[512 * 32];     // [n][k] (transposed)

  const int tid  = threadIdx.x;
  const int lane = tid & 31;
  const int wave = tid >> 5;
  const int m0  = blockIdx.x * 16;
  const int nb0 = blockIdx.y * 512;

  // wave32 WMMA operand mappings (CDNA5 ISA 7.12.2)
  const int mA  = lane & 15;            // A row
  const int kbA = (lane >> 4) << 3;     // A k-group base
  const int nB  = lane & 15;            // B col
  const int kbB = (lane >> 4) << 3;     // B k base

  // staging maps
  const int kg8 = (tid >> 6) << 3;      // 0,8,16,24  (k sub-block)
  const int ng8 = (tid & 63) << 3;      // 0..504     (n sub-block)

  v8f acc[4] = {};

  for (int k0 = 0; k0 < K; k0 += 32) {
    // ---- stage A tile (16x32), 64 b128 loads by first 64 threads ----
    if (tid < 64) {
      const int row = tid >> 2, kv = (tid & 3) << 3;
      *(v8bf*)&ldsA[row * 32 + kv] = ld8(A + (size_t)(m0 + row) * K + k0 + kv);
    }
    // ---- stage B tile transposed via 8x8 register transpose ----
    {
      v8bf r[8];
#pragma unroll
      for (int i = 0; i < 8; ++i)
        r[i] = ld8(Bm + (size_t)(k0 + kg8 + i) * N + nb0 + ng8);
      if (k0 + 32 < K)
        __builtin_prefetch(Bm + (size_t)(k0 + 32 + kg8) * N + nb0 + ng8, 0, 1);
#pragma unroll
      for (int e = 0; e < 8; ++e) {
        v8bf w;
#pragma unroll
        for (int i = 0; i < 8; ++i) w[i] = r[i][e];
        *(v8bf*)&ldsB[(ng8 + e) * 32 + kg8] = w;
      }
    }
    __syncthreads();

    const __bf16* ap = &ldsA[mA * 32 + kbA];
    const v16bf af = cat8(*(const v8bf*)ap, *(const v8bf*)(ap + 16));
#pragma unroll
    for (int t = 0; t < 4; ++t) {
      const __bf16* bp = &ldsB[(wave * 64 + t * 16 + nB) * 32 + kbB];
      const v16bf bfx = cat8(*(const v8bf*)bp, *(const v8bf*)(bp + 16));
      acc[t] = __builtin_amdgcn_wmma_f32_16x16x32_bf16(false, af, false, bfx,
                                                       (short)0, acc[t], false, false);
    }
    __syncthreads();
  }

  const int n0 = nb0 + wave * 64;
  const int rbase = (lane >> 4) << 3;   // C/D: row = r + 8*(lane>>4)
#pragma unroll
  for (int t = 0; t < 4; ++t) {
#pragma unroll
    for (int r = 0; r < 8; ++r) {
      const int row = m0 + rbase + r;
      const int col = n0 + t * 16 + nB;
      float v = acc[t][r] + bias[col];
      if (mode == 0) {
        if (res) v += res[(size_t)row * N + col];
        Cf[(size_t)row * N + col] = v;
      } else if (mode == 1) {
        v = 0.5f * v * (1.0f + erff(v * 0.70710678118f));   // exact GELU
        Cb[(size_t)row * N + col] = f2bf(v);
      } else if (mode == 2) {
        Cb[(size_t)row * N + col] = f2bf(v);
      } else {
        const int bb = row / S_DIM, ss = row % S_DIM;
        const int hh = col / HD_DIM, dd = col % HD_DIM;
        Cb[(((size_t)bb * H_DIM + hh) * S_DIM + ss) * HD_DIM + dd] = f2bf(v * scale);
      }
    }
  }
}

// ---------- flash attention: one wave32 per (b,h, 16-query tile) ----------
// Q pre-scaled by 1/sqrt(HD). Online softmax over 32-key chunks.
// Q/K fragments: contiguous b128 global loads. V staged transposed in LDS.
// P goes D-layout -> LDS -> A-layout (contiguous ds_load_b128 reads).
__global__ void flash_attn_wmma(const __bf16* __restrict__ Q, const __bf16* __restrict__ Kc,
                                const __bf16* __restrict__ V,
                                const int* __restrict__ adj, const float* __restrict__ edge,
                                __bf16* __restrict__ O) {
  const int lane = threadIdx.x & 31;
  const int bh = blockIdx.y;
  const int b = bh / H_DIM, h = bh % H_DIM;
  const int q0 = blockIdx.x * 16;

  const __bf16* qb = Q  + (size_t)bh * S_DIM * HD_DIM;
  const __bf16* kb = Kc + (size_t)bh * S_DIM * HD_DIM;
  const __bf16* vb = V  + (size_t)bh * S_DIM * HD_DIM;
  const int*   adjb  = adj  + (size_t)b * S_DIM * S_DIM;
  const float* edgeb = edge + (size_t)b * S_DIM * S_DIM;

  const int mA  = lane & 15, kbA = (lane >> 4) << 3;
  const int nBl = lane & 15, kbB = (lane >> 4) << 3;
  const int rbase = (lane >> 4) << 3;
  const int vkg8 = (lane >> 3) << 3;    // V staging: k sub-block 0,8,16,24
  const int vdg8 = (lane & 7) << 3;     // V staging: d sub-block 0..56

  __shared__ __align__(16) __bf16 pbuf[16 * 32];   // P row-major [q][k]
  __shared__ __align__(16) __bf16 ldsV[64 * 32];   // V transposed [d][key]

  // Q A-fragments: 16x64 = two 16x32 tiles (contiguous vector loads)
  const __bf16* qrow = qb + (size_t)(q0 + mA) * HD_DIM;
  const v16bf qa0 = cat8(ld8(qrow + kbA),      ld8(qrow + kbA + 16));
  const v16bf qa1 = cat8(ld8(qrow + 32 + kbA), ld8(qrow + 32 + kbA + 16));

  v8f oacc[4] = {};
  float mrow[8], lrow[8];
#pragma unroll
  for (int r = 0; r < 8; ++r) { mrow[r] = -1e30f; lrow[r] = 0.f; }

  for (int kc = 0; kc < S_DIM; kc += 32) {
    if (kc + 32 < S_DIM) {            // prefetch next chunk: one 128B row per lane
      __builtin_prefetch(kb + (size_t)(kc + 32 + lane) * HD_DIM, 0, 1);
      __builtin_prefetch(vb + (size_t)(kc + 32 + lane) * HD_DIM, 0, 1);
    }

    // ---- stage V chunk transposed: [d][key], 8x8 register transpose ----
    {
      v8bf r[8];
#pragma unroll
      for (int i = 0; i < 8; ++i)
        r[i] = ld8(vb + (size_t)(kc + vkg8 + i) * HD_DIM + vdg8);
#pragma unroll
      for (int e = 0; e < 8; ++e) {
        v8bf w;
#pragma unroll
        for (int i = 0; i < 8; ++i) w[i] = r[i][e];
        *(v8bf*)&ldsV[(vdg8 + e) * 32 + vkg8] = w;
      }
    }

    // ---- scores: 16 queries x 32 keys (K rows are contiguous per lane) ----
    v8f sc[2] = {};
#pragma unroll
    for (int t = 0; t < 2; ++t) {
      const __bf16* krow = kb + (size_t)(kc + t * 16 + nBl) * HD_DIM;  // B(k,n)=K[n][k]
      const v16bf b0 = cat8(ld8(krow + kbB),      ld8(krow + kbB + 16));
      const v16bf b1 = cat8(ld8(krow + 32 + kbB), ld8(krow + 32 + kbB + 16));
      sc[t] = __builtin_amdgcn_wmma_f32_16x16x32_bf16(false, qa0, false, b0, (short)0, sc[t], false, false);
      sc[t] = __builtin_amdgcn_wmma_f32_16x16x32_bf16(false, qa1, false, b1, (short)0, sc[t], false, false);
    }

    // ---- add graph bias, chunk row-max ----
    float cm[8];
#pragma unroll
    for (int r = 0; r < 8; ++r) {
      const int qs = q0 + rbase + r;
      {
        const int ks = kc + nBl;
        const int a = adjb[(size_t)qs * S_DIM + ks];
        sc[0][r] += (a > 0) ? 0.1f * edgeb[(size_t)qs * S_DIM + ks] * (float)a : -1e30f;
      }
      {
        const int ks = kc + 16 + nBl;
        const int a = adjb[(size_t)qs * S_DIM + ks];
        sc[1][r] += (a > 0) ? 0.1f * edgeb[(size_t)qs * S_DIM + ks] * (float)a : -1e30f;
      }
      cm[r] = fmaxf(sc[0][r], sc[1][r]);
    }
#pragma unroll
    for (int off = 1; off < 16; off <<= 1)
#pragma unroll
      for (int r = 0; r < 8; ++r)
        cm[r] = fmaxf(cm[r], __shfl_xor(cm[r], off, 32));

    // ---- online softmax update ----
    float p[2][8], corr[8], rs[8];
#pragma unroll
    for (int r = 0; r < 8; ++r) {
      const float mnew = fmaxf(mrow[r], cm[r]);
      corr[r] = __expf(mrow[r] - mnew);
      mrow[r] = mnew;
      p[0][r] = __expf(sc[0][r] - mnew);
      p[1][r] = __expf(sc[1][r] - mnew);
      rs[r] = p[0][r] + p[1][r];
    }
#pragma unroll
    for (int off = 1; off < 16; off <<= 1)
#pragma unroll
      for (int r = 0; r < 8; ++r)
        rs[r] += __shfl_xor(rs[r], off, 32);
#pragma unroll
    for (int r = 0; r < 8; ++r) {
      lrow[r] = lrow[r] * corr[r] + rs[r];
#pragma unroll
      for (int t = 0; t < 4; ++t) oacc[t][r] *= corr[r];
    }

    // ---- P: D-layout -> LDS row-major [q][k] ----
#pragma unroll
    for (int t = 0; t < 2; ++t)
#pragma unroll
      for (int r = 0; r < 8; ++r)
        pbuf[(rbase + r) * 32 + t * 16 + nBl] = f2bf(p[t][r]);
    asm volatile("s_wait_dscnt 0" ::: "memory");   // single wave: LDS in-order, drain stores

    // P as A-fragment (contiguous ds_load_b128 reads)
    const __bf16* pp = &pbuf[mA * 32 + kbA];
    const v16bf pa = cat8(*(const v8bf*)pp, *(const v8bf*)(pp + 16));

    // ---- O += P @ V (K=32 keys, N=64 dims), V from transposed LDS ----
#pragma unroll
    for (int t = 0; t < 4; ++t) {
      const __bf16* vp = &ldsV[(t * 16 + nBl) * 32 + kbB];
      const v16bf vf = cat8(*(const v8bf*)vp, *(const v8bf*)(vp + 16));
      oacc[t] = __builtin_amdgcn_wmma_f32_16x16x32_bf16(false, pa, false, vf, (short)0, oacc[t], false, false);
    }
    asm volatile("s_wait_dscnt 0" ::: "memory");   // drain reads before next chunk's stores
  }

  // ---- epilogue: normalize, store [B,S,D] bf16 ----
#pragma unroll
  for (int t = 0; t < 4; ++t)
#pragma unroll
    for (int r = 0; r < 8; ++r) {
      const int qs = q0 + rbase + r;
      O[((size_t)b * S_DIM + qs) * D_DIM + h * HD_DIM + t * 16 + nBl] =
          f2bf(oacc[t][r] / lrow[r]);
    }
}

// ---------------------------------------------------------------------------
extern "C" void kernel_launch(void* const* d_in, const int* in_sizes, int n_in,
                              void* d_out, int out_size, void* d_ws, size_t ws_size,
                              hipStream_t stream) {
  (void)in_sizes; (void)n_in; (void)out_size; (void)ws_size;
  const float* x    = (const float*)d_in[0];
  const int*   adj  = (const int*)  d_in[1];
  const float* edge = (const float*)d_in[2];
  const float* ln1g = (const float*)d_in[3];
  const float* ln1b = (const float*)d_in[4];
  const float* ln2g = (const float*)d_in[5];
  const float* ln2b = (const float*)d_in[6];
  const float* wq = (const float*)d_in[7];   const float* bq = (const float*)d_in[8];
  const float* wk = (const float*)d_in[9];   const float* bk = (const float*)d_in[10];
  const float* wv = (const float*)d_in[11];  const float* bv = (const float*)d_in[12];
  const float* wo = (const float*)d_in[13];  const float* bo = (const float*)d_in[14];
  const float* w1 = (const float*)d_in[15];  const float* b1 = (const float*)d_in[16];
  const float* w2 = (const float*)d_in[17];  const float* b2 = (const float*)d_in[18];
  float* out = (float*)d_out;

  const int M = B_DIM * S_DIM;  // 4096 rows

  char* ws = (char*)d_ws;
  size_t off = 0;
  auto alloc = [&](size_t bytes) -> char* {
    char* p = ws + off;
    off += (bytes + 255) & ~(size_t)255;
    return p;
  };
  __bf16* nxb = (__bf16*)alloc((size_t)M * D_DIM * 2);       // LN1 out
  __bf16* qhb = (__bf16*)alloc((size_t)M * D_DIM * 2);       // Q [B,H,S,HD] (pre-scaled)
  __bf16* khb = (__bf16*)alloc((size_t)M * D_DIM * 2);       // K [B,H,S,HD]
  __bf16* vhb = (__bf16*)alloc((size_t)M * D_DIM * 2);       // V [B,H,S,HD]
  __bf16* aob = (__bf16*)alloc((size_t)M * D_DIM * 2);       // attention out [B,S,D]
  float*  x1  = (float*) alloc((size_t)M * D_DIM * 4);       // residual 1
  __bf16* n2b = (__bf16*)alloc((size_t)M * D_DIM * 2);       // LN2 out
  __bf16* hb  = (__bf16*)alloc((size_t)M * D4_DIM * 2);      // MLP hidden (post-GELU)
  __bf16* wqb = (__bf16*)alloc((size_t)D_DIM * D_DIM * 2);
  __bf16* wkb = (__bf16*)alloc((size_t)D_DIM * D_DIM * 2);
  __bf16* wvb = (__bf16*)alloc((size_t)D_DIM * D_DIM * 2);
  __bf16* wob = (__bf16*)alloc((size_t)D_DIM * D_DIM * 2);
  __bf16* w1b = (__bf16*)alloc((size_t)D_DIM * D4_DIM * 2);
  __bf16* w2b = (__bf16*)alloc((size_t)D4_DIM * D_DIM * 2);

  // 1) weight conversion
  const int nWs = D_DIM * D_DIM, nWb = D_DIM * D4_DIM;
  cvt_f32_bf16<<<512, 256, 0, stream>>>(wq, wqb, nWs);
  cvt_f32_bf16<<<512, 256, 0, stream>>>(wk, wkb, nWs);
  cvt_f32_bf16<<<512, 256, 0, stream>>>(wv, wvb, nWs);
  cvt_f32_bf16<<<512, 256, 0, stream>>>(wo, wob, nWs);
  cvt_f32_bf16<<<2048, 256, 0, stream>>>(w1, w1b, nWb);
  cvt_f32_bf16<<<2048, 256, 0, stream>>>(w2, w2b, nWb);

  // 2) LN1
  layernorm_to_bf16<<<M, 128, 0, stream>>>(x, ln1g, ln1b, nxb);

  // 3) Q/K/V projections (Q pre-scaled by 1/sqrt(HD)=0.125) into head layout
  dim3 gProj(M / 16, 1);
  gemm_bf16_wmma<<<gProj, 256, 0, stream>>>(nxb, wqb, bq, nullptr, qhb, nullptr,
                                            M, D_DIM, D_DIM, 3, 0.125f);
  gemm_bf16_wmma<<<gProj, 256, 0, stream>>>(nxb, wkb, bk, nullptr, khb, nullptr,
                                            M, D_DIM, D_DIM, 3, 1.0f);
  gemm_bf16_wmma<<<gProj, 256, 0, stream>>>(nxb, wvb, bv, nullptr, vhb, nullptr,
                                            M, D_DIM, D_DIM, 3, 1.0f);

  // 4) fused masked flash attention
  flash_attn_wmma<<<dim3(S_DIM / 16, B_DIM * H_DIM), 32, 0, stream>>>(
      qhb, khb, vhb, adj, edge, aob);

  // 5) output projection + residual -> x1 (fp32)
  gemm_bf16_wmma<<<gProj, 256, 0, stream>>>(aob, wob, bo, x1, nullptr, x,
                                            M, D_DIM, D_DIM, 0, 1.0f);

  // 6) LN2
  layernorm_to_bf16<<<M, 128, 0, stream>>>(x1, ln2g, ln2b, n2b);

  // 7) MLP up + exact GELU
  gemm_bf16_wmma<<<dim3(M / 16, D4_DIM / 512), 256, 0, stream>>>(
      n2b, w1b, b1, nullptr, hb, nullptr, M, D4_DIM, D_DIM, 1, 1.0f);

  // 8) MLP down + residual -> d_out (fp32)
  gemm_bf16_wmma<<<gProj, 256, 0, stream>>>(hb, w2b, b2, out, nullptr, x1,
                                            M, D_DIM, D4_DIM, 0, 1.0f);
}
